// Permu_Gaussian_NN_Htanh_71279277245134
// MI455X (gfx1250) — matrix operand order repaired and checked by reading, not compiled, not run on hardware
//
#include <hip/hip_runtime.h>

// Problem constants (match reference)
constexpr int kJ = 64, kXD = 512, kHD = 1024, kYD = 16, kP = 8, kN = 1024;

constexpr int kWaves = 8;           // 256 threads, 128 rows per WG
constexpr int kRows  = kWaves * 16;
constexpr int kWS    = 520;         // padded K-stride of W1 LDS panel (elements)
constexpr int kW2S   = 40;          // padded K-stride of W2 LDS panel (elements)

typedef __bf16 bf16;
typedef bf16  v16bf __attribute__((ext_vector_type(16)));
typedef bf16  v4bf  __attribute__((ext_vector_type(4)));
typedef float v8f   __attribute__((ext_vector_type(8)));

__device__ __forceinline__ float htanh(float v) {
  return fminf(fmaxf(v, -1.0f), 1.0f);
}

// blockIdx.x = n-tile (128 rows), blockIdx.y = j. Wave w owns rows [n0+16w, n0+16w+16).
__global__ __launch_bounds__(256)
void permu_gaussian_nn_htanh_kernel(
    const float* __restrict__ x,
    const float* __restrict__ muW1, const float* __restrict__ mub1,
    const float* __restrict__ muW2, const float* __restrict__ mub2,
    const float* __restrict__ sgW1, const float* __restrict__ sgb1,
    const float* __restrict__ sgW2, const float* __restrict__ sgb2,
    const float* __restrict__ eW1,  const float* __restrict__ eb1,
    const float* __restrict__ eW2,  const float* __restrict__ eb2,
    const float* __restrict__ permu,
    float* __restrict__ out)
{
  // W1 panel: [32 cols][512 K] bf16, col-major with padded stride (33.3 KB)
  __shared__ __align__(16) bf16 wlds[32 * kWS];
  // W2 panel for current chunk: [16 yd][32 K] bf16, padded stride (1.25 KB)
  __shared__ __align__(16) bf16 w2lds[16 * kW2S];
  // per-wave h-tile staging for C-layout -> A-layout transpose (8 KB)
  __shared__ __align__(16) bf16 hstage[kWaves][16 * 32];

  const int tid   = threadIdx.x;
  const int wave  = tid >> 5;
  const int lane  = tid & 31;
  const int lhalf = lane >> 4;   // 0: lanes 0-15, 1: lanes 16-31
  const int lmod  = lane & 15;

  const int n0 = blockIdx.x * kRows;
  const int j  = blockIdx.y;

  const size_t eW1base = (size_t)j * kXD * kHD;
  const size_t eW2base = (size_t)j * kHD * kYD;

  // ---- Build all 16 A-fragments (this wave's 16 rows x K=512) in registers ----
  // A fragments use contiguous K order (same as B): lane = row (lmod); element e -> K = e + lhalf*16
  v16bf Areg[16];
  {
    const float* xrow = x + (size_t)(n0 + wave * 16 + lmod) * kXD;
#pragma unroll
    for (int kt = 0; kt < 16; ++kt) {
      const int kb = kt * 32 + lhalf * 16;
      float4 a0 = *(const float4*)(xrow + kb);
      float4 a1 = *(const float4*)(xrow + kb + 4);
      float4 a2 = *(const float4*)(xrow + kb + 8);
      float4 a3 = *(const float4*)(xrow + kb + 12);
      v16bf A;
      A[0]  = (bf16)a0.x; A[1]  = (bf16)a0.y; A[2]  = (bf16)a0.z; A[3]  = (bf16)a0.w;
      A[4]  = (bf16)a1.x; A[5]  = (bf16)a1.y; A[6]  = (bf16)a1.z; A[7]  = (bf16)a1.w;
      A[8]  = (bf16)a2.x; A[9]  = (bf16)a2.y; A[10] = (bf16)a2.z; A[11] = (bf16)a2.w;
      A[12] = (bf16)a3.x; A[13] = (bf16)a3.y; A[14] = (bf16)a3.z; A[15] = (bf16)a3.w;
      Areg[kt] = A;
    }
  }

  v8f oacc = {0.f, 0.f, 0.f, 0.f, 0.f, 0.f, 0.f, 0.f};   // GEMM2 C/D tile [16 x 16]

  // Staging roles (256 threads): W1: thread covers 4 K-rows x 16 cols.
  const int kb4 = (tid & 127) * 4;     // K base (4 contiguous K per thread)
  const int ch  = (tid >> 7) * 16;     // column half (0 or 16)

  for (int cc = 0; cc < kHD / 32; ++cc) {   // 32 hidden columns per chunk
    const int hd0 = cc * 32;
    __syncthreads();   // previous chunk's wlds/w2lds reads complete

    // ---- stage W1 = mu + e*sig, [512 K][32 cols] -> wlds[col][K], b64-packed ----
    {
      const size_t wbase = (size_t)kb4 * kHD + hd0 + ch;
#pragma unroll
      for (int q = 0; q < 4; ++q) {            // 4 column groups of 4
        float4 wk[4];
#pragma unroll
        for (int ki = 0; ki < 4; ++ki) {       // 4 contiguous K rows
          const size_t off = wbase + (size_t)ki * kHD + q * 4;
          float4 e4 = *(const float4*)(eW1 + eW1base + off);
          float4 m4 = *(const float4*)(muW1 + off);
          float4 s4 = *(const float4*)(sgW1 + off);
          wk[ki].x = m4.x + e4.x * s4.x;
          wk[ki].y = m4.y + e4.y * s4.y;
          wk[ki].z = m4.z + e4.z * s4.z;
          wk[ki].w = m4.w + e4.w * s4.w;
        }
        bf16* wp = wlds + (ch + q * 4) * kWS + kb4;
        v4bf p0 = {(bf16)wk[0].x, (bf16)wk[1].x, (bf16)wk[2].x, (bf16)wk[3].x};
        v4bf p1 = {(bf16)wk[0].y, (bf16)wk[1].y, (bf16)wk[2].y, (bf16)wk[3].y};
        v4bf p2 = {(bf16)wk[0].z, (bf16)wk[1].z, (bf16)wk[2].z, (bf16)wk[3].z};
        v4bf p3 = {(bf16)wk[0].w, (bf16)wk[1].w, (bf16)wk[2].w, (bf16)wk[3].w};
        *(v4bf*)(wp)            = p0;   // col ch+4q+0
        *(v4bf*)(wp + kWS)      = p1;   // col ch+4q+1
        *(v4bf*)(wp + 2 * kWS)  = p2;   // col ch+4q+2
        *(v4bf*)(wp + 3 * kWS)  = p3;   // col ch+4q+3
      }
    }
    // ---- stage W2 chunk [32 K][16 yd] -> w2lds[yd][K] (2 elements/thread) ----
    {
      const int k  = tid >> 3;          // 0..31
      const int c0 = (tid & 7) * 2;     // 0,2,..,14
      const size_t idx = (size_t)(hd0 + k) * kYD + c0;
      float2 m2 = *(const float2*)(muW2 + idx);
      float2 s2 = *(const float2*)(sgW2 + idx);
      float2 e2 = *(const float2*)(eW2 + eW2base + idx);
      w2lds[c0 * kW2S + k]       = (bf16)(m2.x + e2.x * s2.x);
      w2lds[(c0 + 1) * kW2S + k] = (bf16)(m2.y + e2.y * s2.y);
    }
    __syncthreads();

    // ---- GEMM1: 16 ktiles x 2 column tiles, A from registers, B from LDS ----
    v8f acc0 = {0.f, 0.f, 0.f, 0.f, 0.f, 0.f, 0.f, 0.f}; // cols hd0..hd0+15
    v8f acc1 = {0.f, 0.f, 0.f, 0.f, 0.f, 0.f, 0.f, 0.f}; // cols hd0+16..hd0+31
    {
      // B layout (32x16): lane = col, K = kt*32 + lhalf*16 + 0..15 (contiguous)
      const bf16* b0base = wlds + lmod * kWS + lhalf * 16;
      const bf16* b1base = wlds + (16 + lmod) * kWS + lhalf * 16;
#pragma unroll
      for (int kt = 0; kt < 16; ++kt) {
        v16bf B0 = *(const v16bf*)(b0base + kt * 32);
        v16bf B1 = *(const v16bf*)(b1base + kt * 32);
        acc0 = __builtin_amdgcn_wmma_f32_16x16x32_bf16(false, Areg[kt], false, B0,
                                                       (short)0, acc0, false, false);
        acc1 = __builtin_amdgcn_wmma_f32_16x16x32_bf16(false, Areg[kt], false, B1,
                                                       (short)0, acc1, false, false);
      }
    }

    // ---- bias + hardtanh; stage h tile (C-layout -> A-layout, wave-private) ----
    const float b1v0 = mub1[hd0 + lmod] +
                       eb1[(size_t)j * kHD + hd0 + lmod] * sgb1[hd0 + lmod];
    const float b1v1 = mub1[hd0 + 16 + lmod] +
                       eb1[(size_t)j * kHD + hd0 + 16 + lmod] * sgb1[hd0 + 16 + lmod];
    bf16* hs = hstage[wave];
#pragma unroll
    for (int r = 0; r < 8; ++r) {
      const int row = r + lhalf * 8;                  // C/D layout row
      hs[row * 32 + lmod]      = (bf16)htanh(acc0[r] + b1v0);
      hs[row * 32 + 16 + lmod] = (bf16)htanh(acc1[r] + b1v1);
    }
    // wave-private region; DS ops of one wave complete in order

    // A fragment of h (16 rows x K=32 hidden cols)
    const bf16* ha = hs + lmod * 32 + lhalf * 16;
    v16bf Ah;
    {
      v16bf lo_hi;
#pragma unroll
      for (int e = 0; e < 8; ++e) lo_hi[e] = ha[e];
#pragma unroll
      for (int e = 0; e < 8; ++e) lo_hi[8 + e] = ha[8 + e];
      Ah = lo_hi;
    }
    // W2 B fragment: lane = yd col, K contiguous from padded panel
    v16bf Bw = *(const v16bf*)(w2lds + lmod * kW2S + lhalf * 16);

    oacc = __builtin_amdgcn_wmma_f32_16x16x32_bf16(false, Ah, false, Bw,
                                                   (short)0, oacc, false, false);
  }

  // ---- epilogue: + b2, apply 8 permutations (column scatter), store ----
  const float b2v = mub2[lmod] + eb2[(size_t)j * kYD + lmod] * sgb2[lmod];

  int ydst[kP];
#pragma unroll
  for (int p = 0; p < kP; ++p) {
    const float* pm = permu + (size_t)p * kYD * kYD + lmod * kYD;  // row c=lmod of permu[p]
    int y = 0;
#pragma unroll
    for (int q = 0; q < kYD; ++q) y += (pm[q] > 0.5f) ? q : 0;
    ydst[p] = y;
  }

#pragma unroll
  for (int r = 0; r < 8; ++r) {
    const int grow = n0 + wave * 16 + lhalf * 8 + r;
    const float val = oacc[r] + b2v;
#pragma unroll
    for (int p = 0; p < kP; ++p) {
      out[((size_t)(j * kP + p) * kN + grow) * kYD + ydst[p]] = val;
    }
  }
}

extern "C" void kernel_launch(void* const* d_in, const int* in_sizes, int n_in,
                              void* d_out, int out_size, void* d_ws, size_t ws_size,
                              hipStream_t stream) {
  const float* x    = (const float*)d_in[0];
  const float* muW1 = (const float*)d_in[1];
  const float* mub1 = (const float*)d_in[2];
  const float* muW2 = (const float*)d_in[3];
  const float* mub2 = (const float*)d_in[4];
  const float* sgW1 = (const float*)d_in[5];
  const float* sgb1 = (const float*)d_in[6];
  const float* sgW2 = (const float*)d_in[7];
  const float* sgb2 = (const float*)d_in[8];
  const float* eW1  = (const float*)d_in[9];
  const float* eb1  = (const float*)d_in[10];
  const float* eW2  = (const float*)d_in[11];
  const float* eb2  = (const float*)d_in[12];
  const float* prm  = (const float*)d_in[13];

  dim3 grid(kN / kRows, kJ);   // 8 row-tiles x 64 ensemble members
  permu_gaussian_nn_htanh_kernel<<<grid, 256, 0, stream>>>(
      x, muW1, mub1, muW2, mub2, sgW1, sgb1, sgW2, sgb2,
      eW1, eb1, eW2, eb2, prm, (float*)d_out);
}
